// MyLSTM_30975304138984
// MI455X (gfx1250) — matrix operand order repaired and checked by reading, not compiled, not run on hardware
//
#include <hip/hip_runtime.h>
#include <hip/hip_bf16.h>

// ---------------------------------------------------------------------------
// Bidirectional 2-layer LSTM for MI455X (gfx1250, wave32, WMMA).
//   L=512, B=64, D_IN=256, H=512.
// Strategy:
//  - One persistent kernel launch per layer, 64 workgroups (32 per direction).
//  - Each WG owns 16 hidden units (=> 64 gate rows of W_ih|W_hh), keeps the
//    bf16 weight slice resident in LDS for all 512 timesteps.
//  - Per step: stage [x_t ; h_{t-1}] (64 x (din+512)) into LDS as bf16, run
//    v_wmma_f32_16x16x32_bf16 GEMM (M-tiles=4 waves, N-tiles=4 gates),
//    LSTM cell update in registers (c state persists in VGPRs), write h slice
//    to global, grid-barrier across the 32 WGs of the direction.
//  - Both directions run concurrently (independent barrier counters).
//  - Layer-0 h history stored bf16 in workspace; layer-1 writes f32 into
//    d_out, which also serves as its own recurrence history.
// ---------------------------------------------------------------------------

typedef __attribute__((ext_vector_type(16))) __bf16 v16bf;
typedef __attribute__((ext_vector_type(8)))  __bf16 v8bf;
typedef __attribute__((ext_vector_type(8)))  float  v8f;

union BF16x16 {
    v16bf v;
    struct { v8bf lo, hi; } h8;
};

struct alignas(8) BF4 { __bf16 x[4]; };

__device__ __forceinline__ void cvt4(__bf16* d, float4 f) {
    d[0] = (__bf16)f.x; d[1] = (__bf16)f.y; d[2] = (__bf16)f.z; d[3] = (__bf16)f.w;
}
__device__ __forceinline__ float sigm_f(float x)  { return 1.0f / (1.0f + __expf(-x)); }
__device__ __forceinline__ float tanh_fast(float x){ return 1.0f - 2.0f / (__expf(2.0f * x) + 1.0f); }

struct RecArgs {
    const void*  x[2];      // per-direction input sequence base (f32 or bf16)
    const float* wih[2];    // [4H, din] f32
    const float* whh[2];    // [4H, H]   f32
    const float* bih[2];    // [4H]
    const float* bhh[2];    // [4H]
    void*        h[2];      // per-direction h history (written + read back)
    unsigned*    counters;  // [2] grid-barrier counters (one per direction)
    int ldx, din, ldh;
    int oh[2];              // column offset into h rows per direction
};

// XB: x source is bf16 (else f32).  HB: h history is bf16 (else f32).
template <bool XB, bool HB>
__global__ __launch_bounds__(128, 1) void lstm_rec_kernel(RecArgs args) {
    constexpr int Hd = 512, Bsz = 64, Lt = 512, NWG = 32;
    const int dir  = (int)(blockIdx.x >> 5);   // 0 = forward, 1 = backward
    const int wg   = (int)(blockIdx.x & 31);
    const int j0   = wg * 16;                  // first hidden unit of slice
    const int tid  = (int)threadIdx.x;
    const int wave = tid >> 5;
    const int lane = tid & 31;
    const int u    = lane & 15;                // column within 16-wide tiles
    const int din  = args.din;
    const int KS   = din + Hd + 8;             // padded LDS row stride (elems)

    extern __shared__ char smem[];
    __bf16* wlds = (__bf16*)smem;              // [64][KS] weight slice
    __bf16* alds = wlds + 64 * KS;             // [64][KS] A = [x_t ; h_prev]

    // ---- load + convert the weight slice (rows interleaved gate*16 + unit) --
    {
        const float* wih = args.wih[dir];
        const float* whh = args.whh[dir];
        for (int r = wave; r < 64; r += 4) {
            const int g = r >> 4, uu = r & 15;
            const float* wi = wih + (size_t)(g * Hd + j0 + uu) * din;
            const float* wh = whh + (size_t)(g * Hd + j0 + uu) * Hd;
            __bf16* dst = wlds + (size_t)r * KS;
            for (int c = lane * 4; c < din; c += 128)
                cvt4(dst + c, *(const float4*)(wi + c));
            for (int c = lane * 4; c < Hd; c += 128)
                cvt4(dst + din + c, *(const float4*)(wh + c));
        }
    }

    // ---- per-lane biases: column u of each gate tile -----------------------
    float bias[4];
#pragma unroll
    for (int g = 0; g < 4; ++g)
        bias[g] = args.bih[dir][g * Hd + j0 + u] + args.bhh[dir][g * Hd + j0 + u];

    // cell state: 8 (batch,unit) pairs per lane, persists across all steps
    float cst[8];
#pragma unroll
    for (int v = 0; v < 8; ++v) cst[v] = 0.0f;

    unsigned* cnt   = args.counters + dir;
    const int mt    = wave;                     // M tile = batch rows 16*mt..
    const int koffA = (lane >= 16) ? 16 : 0;    // A fragment K-half select
    const int koffB = (lane >= 16) ? 16 : 0;    // B fragment K-half select
    const int hiM   = (lane >= 16) ? 8 : 0;     // C/D M offset for hi lanes
    const int ldx   = args.ldx, ldh = args.ldh, oh = args.oh[dir];
    const int Ktot  = din + Hd;
    const __bf16* arow = alds + (size_t)(mt * 16 + u) * KS;

    __syncthreads();

    for (int step = 0; step < Lt; ++step) {
        const int t  = dir ? (Lt - 1 - step) : step;
        const int tp = dir ? (t + 1) : (t - 1);

        // ---- stage A = [x_t ; h_{t-1}] into LDS as bf16 --------------------
        for (int r = wave; r < 64; r += 4) {
            __bf16* dst = alds + (size_t)r * KS;
            if (XB) {
                const __bf16* xs = (const __bf16*)args.x[dir] + ((size_t)t * Bsz + r) * ldx;
                for (int c = lane * 4; c < din; c += 128)
                    *(BF4*)(dst + c) = *(const BF4*)(xs + c);
            } else {
                const float* xs = (const float*)args.x[dir] + ((size_t)t * Bsz + r) * ldx;
                for (int c = lane * 4; c < din; c += 128)
                    cvt4(dst + c, *(const float4*)(xs + c));
            }
            if (step == 0) {
                BF4 z; z.x[0] = z.x[1] = z.x[2] = z.x[3] = (__bf16)0.0f;
                for (int c = lane * 4; c < Hd; c += 128)
                    *(BF4*)(dst + din + c) = z;
            } else if (HB) {
                const __bf16* hs = (const __bf16*)args.h[dir] + ((size_t)tp * Bsz + r) * ldh + oh;
                for (int c = lane * 4; c < Hd; c += 128)
                    *(BF4*)(dst + din + c) = *(const BF4*)(hs + c);
            } else {
                const float* hs = (const float*)args.h[dir] + ((size_t)tp * Bsz + r) * ldh + oh;
                for (int c = lane * 4; c < Hd; c += 128)
                    cvt4(dst + din + c, *(const float4*)(hs + c));
            }
        }
        __syncthreads();

        // ---- WMMA GEMM: gates[64B x 64gates] over K = din + H --------------
        v8f acc[4];
#pragma unroll
        for (int nt = 0; nt < 4; ++nt)
#pragma unroll
            for (int v = 0; v < 8; ++v) acc[nt][v] = bias[nt];

        for (int kb = 0; kb < Ktot; kb += 32) {
            BF16x16 a;
            a.h8.lo = *(const v8bf*)(arow + kb + koffA);        // K koff..koff+7
            a.h8.hi = *(const v8bf*)(arow + kb + koffA + 8);    // K koff+8..+15
#pragma unroll
            for (int nt = 0; nt < 4; ++nt) {
                const __bf16* bp = wlds + (size_t)(nt * 16 + u) * KS + kb + koffB;
                BF16x16 b;
                b.h8.lo = *(const v8bf*)bp;                     // K koffB..+7
                b.h8.hi = *(const v8bf*)(bp + 8);               // K koffB+8..+15
                acc[nt] = __builtin_amdgcn_wmma_f32_16x16x32_bf16(
                    false, a.v, false, b.v, (short)0, acc[nt], false, false);
            }
        }

        // ---- LSTM cell update + h write ------------------------------------
#pragma unroll
        for (int v = 0; v < 8; ++v) {
            const float gi = sigm_f(acc[0][v]);
            const float gf = sigm_f(acc[1][v]);
            const float gg = tanh_fast(acc[2][v]);
            const float go = sigm_f(acc[3][v]);
            const float c  = gf * cst[v] + gi * gg;
            cst[v] = c;
            const float hval = go * tanh_fast(c);
            const int b = mt * 16 + hiM + v;
            const size_t off = ((size_t)t * Bsz + b) * ldh + oh + j0 + u;
            if (HB) ((__bf16*)args.h[dir])[off] = (__bf16)hval;
            else    ((float*) args.h[dir])[off] = hval;
        }

        // ---- grid barrier across the 32 WGs of this direction --------------
        __threadfence();
        __syncthreads();
        if (tid == 0) {
            __hip_atomic_fetch_add(cnt, 1u, __ATOMIC_RELEASE, __HIP_MEMORY_SCOPE_AGENT);
            const unsigned target = (unsigned)(NWG * (step + 1));
            while (__hip_atomic_load(cnt, __ATOMIC_ACQUIRE, __HIP_MEMORY_SCOPE_AGENT) < target)
                __builtin_amdgcn_s_sleep(8);
        }
        __syncthreads();
    }
}

__global__ void zero_counters_kernel(unsigned* c) {
    if (threadIdx.x < 8) c[threadIdx.x] = 0u;
}

extern "C" void kernel_launch(void* const* d_in, const int* in_sizes, int n_in,
                              void* d_out, int out_size, void* d_ws, size_t ws_size,
                              hipStream_t stream) {
    (void)in_sizes; (void)n_in; (void)out_size; (void)ws_size;
    constexpr int L = 512, B = 64, D_IN = 256, H = 512;

    const float* inputs = (const float*)d_in[0];
    // dict order: inputs, then f0(w_ih,w_hh,b_ih,b_hh), f1(...), b0(...), b1(...)
    const float* wih_f0 = (const float*)d_in[1];
    const float* whh_f0 = (const float*)d_in[2];
    const float* bih_f0 = (const float*)d_in[3];
    const float* bhh_f0 = (const float*)d_in[4];
    const float* wih_f1 = (const float*)d_in[5];
    const float* whh_f1 = (const float*)d_in[6];
    const float* bih_f1 = (const float*)d_in[7];
    const float* bhh_f1 = (const float*)d_in[8];
    const float* wih_b0 = (const float*)d_in[9];
    const float* whh_b0 = (const float*)d_in[10];
    const float* bih_b0 = (const float*)d_in[11];
    const float* bhh_b0 = (const float*)d_in[12];
    const float* wih_b1 = (const float*)d_in[13];
    const float* whh_b1 = (const float*)d_in[14];
    const float* bih_b1 = (const float*)d_in[15];
    const float* bhh_b1 = (const float*)d_in[16];

    // workspace layout: h0f bf16 (32MB) | h0b bf16 (32MB) | 8 barrier counters
    const size_t hElems = (size_t)L * B * H;
    __bf16* h0f = (__bf16*)d_ws;
    __bf16* h0b = h0f + hElems;
    unsigned* cnts = (unsigned*)((char*)d_ws + 2 * hElems * sizeof(__bf16));

    zero_counters_kernel<<<1, 32, 0, stream>>>(cnts);

    const size_t smem0 = (size_t)2 * 64 * (D_IN + H + 8) * sizeof(__bf16); // ~194KB
    const size_t smem1 = (size_t)2 * 64 * (H + H + 8) * sizeof(__bf16);    // ~258KB
    hipFuncSetAttribute(reinterpret_cast<const void*>(&lstm_rec_kernel<false, true>),
                        hipFuncAttributeMaxDynamicSharedMemorySize, (int)smem0);
    hipFuncSetAttribute(reinterpret_cast<const void*>(&lstm_rec_kernel<true, false>),
                        hipFuncAttributeMaxDynamicSharedMemorySize, (int)smem1);

    // ---- Layer 0: x = inputs (f32), h history stored bf16 in ws -----------
    RecArgs a0{};
    a0.x[0] = inputs;   a0.x[1] = inputs;
    a0.wih[0] = wih_f0; a0.whh[0] = whh_f0; a0.bih[0] = bih_f0; a0.bhh[0] = bhh_f0;
    a0.wih[1] = wih_b0; a0.whh[1] = whh_b0; a0.bih[1] = bih_b0; a0.bhh[1] = bhh_b0;
    a0.h[0] = h0f;      a0.h[1] = h0b;
    a0.counters = cnts;
    a0.ldx = D_IN; a0.din = D_IN; a0.ldh = H;
    a0.oh[0] = 0; a0.oh[1] = 0;
    lstm_rec_kernel<false, true><<<dim3(64), dim3(128), smem0, stream>>>(a0);

    // ---- Layer 1: x = h0 (bf16), h history = d_out (f32, [L,B,2H]) --------
    RecArgs a1{};
    a1.x[0] = h0f;      a1.x[1] = h0b;
    a1.wih[0] = wih_f1; a1.whh[0] = whh_f1; a1.bih[0] = bih_f1; a1.bhh[0] = bhh_f1;
    a1.wih[1] = wih_b1; a1.whh[1] = whh_b1; a1.bih[1] = bih_b1; a1.bhh[1] = bhh_b1;
    a1.h[0] = d_out;    a1.h[1] = d_out;
    a1.counters = cnts + 2;
    a1.ldx = H; a1.din = H; a1.ldh = 2 * H;
    a1.oh[0] = 0; a1.oh[1] = H;
    lstm_rec_kernel<true, false><<<dim3(64), dim3(128), smem1, stream>>>(a1);
}
